// LinearAttention_85341000172336
// MI455X (gfx1250) — matrix operand order repaired and checked by reading, not compiled, not run on hardware
//
#include <hip/hip_runtime.h>

#define S_LEN 4096
#define DIM   128
#define BH    64
#define EPSV  1e-6f

typedef __attribute__((ext_vector_type(16))) __bf16   v16bf;
typedef __attribute__((ext_vector_type(8)))  float    v8f;
typedef __attribute__((ext_vector_type(4)))  unsigned v4u;
typedef __attribute__((ext_vector_type(8)))  unsigned v8u;

#if __has_builtin(__builtin_amdgcn_sched_barrier)
#define SCHED_FENCE() __builtin_amdgcn_sched_barrier(0)
#else
#define SCHED_FENCE()
#endif

__device__ __forceinline__ float elu1(float x) {
  // elu(x)+1 : x>0 ? x+1 : exp(x)
  return x > 0.0f ? x + 1.0f : __expf(x);
}

// Build a v16bf WMMA fragment from two 16-byte aligned LDS runs.
__device__ __forceinline__ v16bf load_frag_pair(const __bf16* p0, const __bf16* p1) {
  v4u lo = *(const v4u*)p0;
  v4u hi = *(const v4u*)p1;
  v8u w  = __builtin_shufflevector(lo, hi, 0, 1, 2, 3, 4, 5, 6, 7);
  return __builtin_bit_cast(v16bf, w);
}

// LDS row strides: 136 bf16 = 272 B -> rows 16-B aligned (b128 legal),
// 272 B = 68 dwords == 4 (mod 64) -> each 16-lane half sweeps all banks.
#define BSTR 136

// A fragment (16x32 bf16, ISA layout): row r, k-base = 8*half + ks.
#define LOAD_A(T, r, ks) \
  load_frag_pair(&T[r][8 * half + (ks)], &T[r][8 * half + (ks) + 16])
// B fragment (32x16 bf16, tile stored n-major): col n, k-base = 8*half + ks.
#define LOAD_B(T, n, ks) \
  load_frag_pair(&T[n][8 * half + (ks)], &T[n][8 * half + (ks) + 16])

// ---------------------------------------------------------------------------
// Kernel 1: per (bh, s-chunk) split-K state update.
//   S_part[d,m] = sum_{s in chunk} kf[s,d] * vf[s,m]   (bf16 WMMA, f32 acc)
//   ksum_part[d] = sum_s kf[s,d]
//   z_t[d,s] = z_prev[d] + kf[s,d]                     (fp32, coalesced along s)
// ---------------------------------------------------------------------------
__global__ __launch_bounds__(256) void lin_attn_state(
    const float* __restrict__ kg, const float* __restrict__ vg,
    const float* __restrict__ mask, const float* __restrict__ z_prev,
    float* __restrict__ z_out, float* __restrict__ spart,
    float* __restrict__ kpart)
{
  const int tid    = threadIdx.x;
  const int lane   = tid & 31, wv = tid >> 5;
  const int half   = lane >> 4, lanelo = lane & 15;
  const int bh     = blockIdx.y, chunk = blockIdx.x, nchunk = gridDim.x;
  const int s_per_chunk = S_LEN / nchunk;
  const int sbase  = chunk * s_per_chunk;

  extern __shared__ char smem[];
  float  (*kTf)[129]  = (float  (*)[129])(smem);                     // 66048 B
  __bf16 (*kTb)[BSTR] = (__bf16 (*)[BSTR])(smem + 66048);            // 34816 B
  __bf16 (*vtT)[BSTR] = (__bf16 (*)[BSTR])(smem + 66048 + 34816);    // 34816 B

  float ksum_acc = 0.0f;
  v8f acc[8] = {};   // wave wv owns d-rows [wv*16, wv*16+16), all 8 m-tiles

  for (int st = 0; st < s_per_chunk; st += 128) {
    const int s0 = sbase + st;
    __syncthreads();   // previous iteration's LDS readers done
    // Load + feature map. Coalesced along d/m; transpose into LDS.
    for (int i = tid; i < 128 * 128; i += 256) {
      int srow = i >> 7, d = i & 127;
      size_t g = ((size_t)bh * S_LEN + (s0 + srow)) * DIM + d;
      float mk = mask[(size_t)bh * S_LEN + (s0 + srow)];
      float kf = elu1(kg[g]) * mk;
      float vf = vg[g] * mk;
      kTf[d][srow] = kf;
      kTb[d][srow] = (__bf16)kf;
      vtT[d][srow] = (__bf16)vf;   // here d indexes m for the v tensor
    }
    if (st + 128 < s_per_chunk) {
      __builtin_prefetch(&kg[((size_t)bh * S_LEN + (s0 + 128)) * DIM + tid], 0, 1);
      __builtin_prefetch(&vg[((size_t)bh * S_LEN + (s0 + 128)) * DIM + tid], 0, 1);
    }
    __syncthreads();
    // z_t write, coalesced along s (from transposed fp32 tile)
    for (int i = tid; i < 128 * 128; i += 256) {
      int d = i >> 7, j = i & 127;
      z_out[((size_t)bh * DIM + d) * S_LEN + (s0 + j)] =
          z_prev[bh * DIM + d] + kTf[d][j];
    }
    // ksum partial (row sums of fp32 tile)
    if (tid < 128) {
      float s = 0.0f;
      #pragma unroll 8
      for (int j = 0; j < 128; ++j) s += kTf[tid][j];
      ksum_acc += s;
    }
    // WMMA, software-pipelined; sched_barrier pins "loads for n+1 before
    // WMMA n" so each WMMA waits with dscnt<=2 instead of 0.
    {
      const int dA = wv * 16 + lanelo;
      v16bf a_cur = LOAD_A(kTb, dA, 0);
      v16bf b_cur = LOAD_B(vtT, lanelo, 0);
      #pragma unroll
      for (int ks = 0; ks < 128; ks += 32) {
        v16bf a_next;
        if (ks < 96) a_next = LOAD_A(kTb, dA, ks + 32);
        #pragma unroll
        for (int nt = 0; nt < 8; ++nt) {
          v16bf b_next;
          if (nt < 7)
            b_next = LOAD_B(vtT, (nt + 1) * 16 + lanelo, ks);
          else if (ks < 96)
            b_next = LOAD_B(vtT, lanelo, ks + 32);
          SCHED_FENCE();
          acc[nt] = __builtin_amdgcn_wmma_f32_16x16x32_bf16(
              false, a_cur, false, b_cur, (short)0, acc[nt], false, false);
          SCHED_FENCE();
          if (nt < 7 || ks < 96) b_cur = b_next;
        }
        if (ks < 96) a_cur = a_next;
      }
    }
  }

  // Write split-K partials. C/D layout: vgpr v -> M=v+8*half, col = lanelo.
  #pragma unroll
  for (int nt = 0; nt < 8; ++nt)
    #pragma unroll
    for (int vv = 0; vv < 8; ++vv) {
      int d = wv * 16 + vv + 8 * half;
      int m = nt * 16 + lanelo;
      spart[(((size_t)bh * nchunk + chunk) * DIM + d) * DIM + m] = acc[nt][vv];
    }
  if (tid < 128)
    kpart[((size_t)bh * nchunk + chunk) * DIM + tid] = ksum_acc;
}

// ---------------------------------------------------------------------------
// Kernel 2: reduce split-K partials, add S_prev; emit f32 S_t (output),
// TRANSPOSED bf16 S_t copy ([m][d], for kernel 3's n-major B tile) + f32 ksum.
// ---------------------------------------------------------------------------
__global__ __launch_bounds__(256) void lin_attn_reduce(
    const float* __restrict__ spart, const float* __restrict__ kpart,
    const float* __restrict__ S_prev, float* __restrict__ S_out,
    __bf16* __restrict__ sbfT, float* __restrict__ kfin, int nchunk)
{
  __shared__ __bf16 tile[128][129];
  const int bh = blockIdx.x, tid = threadIdx.x;
  for (int i = tid; i < DIM * DIM; i += 256) {
    float s = S_prev[(size_t)bh * DIM * DIM + i];
    for (int c = 0; c < nchunk; ++c)
      s += spart[((size_t)bh * nchunk + c) * DIM * DIM + i];
    S_out[(size_t)bh * DIM * DIM + i] = s;
    tile[i >> 7][i & 127] = (__bf16)s;      // [d][m]
  }
  if (tid < DIM) {
    float s = 0.0f;
    for (int c = 0; c < nchunk; ++c)
      s += kpart[((size_t)bh * nchunk + c) * DIM + tid];
    kfin[bh * DIM + tid] = s;
  }
  __syncthreads();
  for (int i = tid; i < DIM * DIM; i += 256) {
    int m = i >> 7, d = i & 127;
    sbfT[(size_t)bh * DIM * DIM + (size_t)m * DIM + d] = tile[d][m];
  }
}

// ---------------------------------------------------------------------------
// Kernel 3: out[s,m] = (qf[s,:] @ S_t) / (qf[s,:].ksum + eps)
// ---------------------------------------------------------------------------
__global__ __launch_bounds__(256) void lin_attn_out(
    const float* __restrict__ qg, const __bf16* __restrict__ sbfT,
    const float* __restrict__ kfin, float* __restrict__ out)
{
  const int tid  = threadIdx.x;
  const int lane = tid & 31, wv = tid >> 5;
  const int half = lane >> 4, lanelo = lane & 15;
  const int bh   = blockIdx.y;
  const int s0   = blockIdx.x * 128;

  extern __shared__ char smem[];
  __bf16 (*qb )[BSTR] = (__bf16 (*)[BSTR])(smem);            // 34816 B, [s][d] (A)
  __bf16 (*sbT)[BSTR] = (__bf16 (*)[BSTR])(smem + 34816);    // 34816 B, [m][d] (B)
  float* den_s  = (float*)(smem + 34816 + 34816);
  float* ksum_s = den_s + 128;

  if (tid < 128) ksum_s[tid] = kfin[bh * DIM + tid];
  for (int i = tid; i < 128 * 128; i += 256) {
    int srow = i >> 7, d = i & 127;
    qb[srow][d] = (__bf16)elu1(qg[((size_t)bh * S_LEN + (s0 + srow)) * DIM + d]);
  }
  // S_t bf16 already transposed in global -> coalesced read, contiguous LDS rows
  for (int i = tid; i < 128 * 128; i += 256) {
    int m = i >> 7, d = i & 127;
    sbT[m][d] = sbfT[(size_t)bh * DIM * DIM + i];
  }
  __syncthreads();
  if (tid < 128) {
    float den = 0.0f;
    #pragma unroll 8
    for (int d = 0; d < 128; ++d) den += (float)qb[tid][d] * ksum_s[d];
    den_s[tid] = den;
  }
  __syncthreads();

  v8f acc[8] = {};
  {
    const int sA = wv * 16 + lanelo;
    v16bf a_cur = LOAD_A(qb, sA, 0);
    v16bf b_cur = LOAD_B(sbT, lanelo, 0);
    #pragma unroll
    for (int ks = 0; ks < 128; ks += 32) {
      v16bf a_next;
      if (ks < 96) a_next = LOAD_A(qb, sA, ks + 32);
      #pragma unroll
      for (int nt = 0; nt < 8; ++nt) {
        v16bf b_next;
        if (nt < 7)
          b_next = LOAD_B(sbT, (nt + 1) * 16 + lanelo, ks);
        else if (ks < 96)
          b_next = LOAD_B(sbT, lanelo, ks + 32);
        SCHED_FENCE();
        acc[nt] = __builtin_amdgcn_wmma_f32_16x16x32_bf16(
            false, a_cur, false, b_cur, (short)0, acc[nt], false, false);
        SCHED_FENCE();
        if (nt < 7 || ks < 96) b_cur = b_next;
      }
      if (ks < 96) a_cur = a_next;
    }
  }

  #pragma unroll
  for (int nt = 0; nt < 8; ++nt)
    #pragma unroll
    for (int vv = 0; vv < 8; ++vv) {
      int srow = wv * 16 + vv + 8 * half;
      int m    = nt * 16 + lanelo;
      out[((size_t)bh * S_LEN + (s0 + srow)) * DIM + m] =
          acc[nt][vv] / (den_s[srow] + EPSV);
    }
}

// ---------------------------------------------------------------------------
extern "C" void kernel_launch(void* const* d_in, const int* in_sizes, int n_in,
                              void* d_out, int out_size, void* d_ws, size_t ws_size,
                              hipStream_t stream)
{
  const float* q     = (const float*)d_in[0];
  const float* k     = (const float*)d_in[1];
  const float* v     = (const float*)d_in[2];
  const float* mask  = (const float*)d_in[3];
  const float* Sprev = (const float*)d_in[4];
  const float* zprev = (const float*)d_in[5];

  float* out   = (float*)d_out;                        // [64,4096,128]
  float* S_out = out + (size_t)BH * S_LEN * DIM;       // [64,128,128]
  float* z_out = S_out + (size_t)BH * DIM * DIM;       // [64,128,4096]

  // Choose split-K chunk count that fits the workspace.
  auto need = [](int nc) -> size_t {
    return ((size_t)BH * nc * DIM * DIM + (size_t)BH * nc * DIM +
            (size_t)BH * DIM) * sizeof(float) +
           (size_t)BH * DIM * DIM * sizeof(__bf16);
  };
  int nchunk = 8;
  while (nchunk > 1 && need(nchunk) > ws_size) nchunk >>= 1;

  float*  spart = (float*)d_ws;
  float*  kpart = spart + (size_t)BH * nchunk * DIM * DIM;
  float*  kfin  = kpart + (size_t)BH * nchunk * DIM;
  __bf16* sbfT  = (__bf16*)(kfin + (size_t)BH * DIM);

  size_t shm1 = 66048 + 34816 + 34816;           // 135680 B dynamic LDS
  lin_attn_state<<<dim3(nchunk, BH), 256, shm1, stream>>>(
      k, v, mask, zprev, z_out, spart, kpart);

  lin_attn_reduce<<<dim3(BH), 256, 0, stream>>>(
      spart, kpart, Sprev, S_out, sbfT, kfin, nchunk);

  size_t shm3 = 34816 + 34816 + 512 + 512;       // 70656 B dynamic LDS
  lin_attn_out<<<dim3(S_LEN / 128, BH), 256, shm3, stream>>>(
      q, sbfT, kfin, out);
}
